// ConvBlock_57982058496543
// MI455X (gfx1250) — compile-verified, matrix operand order checked by reading, not run on hardware
//
#include <hip/hip_runtime.h>
#include <hip/hip_bf16.h>

// ---------------------------------------------------------------------------
// Deformable conv block for MI455X (gfx1250), wave32 + WMMA f16 (f32 accum).
// Round 2: async global->LDS A-staging (ASYNCcnt path), double-buffered LDS
// (one barrier per K-step), b128 LDS stores for B-staging, branch-free
// precomputed bilinear taps for the deform stage.
// ---------------------------------------------------------------------------

typedef _Float16 h16;
typedef __attribute__((ext_vector_type(16))) _Float16 v16h;
typedef __attribute__((ext_vector_type(8)))  _Float16 v8h;
typedef __attribute__((ext_vector_type(4)))  _Float16 v4h;
typedef __attribute__((ext_vector_type(8)))  float    v8f;

#define Bn 4
#define Cn 96
#define On 96
#define Hn 128
#define Wn 128
#define KKn 9
#define KDIM 864          // KKn * Cn
#define KSTEPS 27         // KDIM / 32
#define LDK 40            // LDS row stride in halves (80B, 16B aligned)
#define NPIX (Hn*Wn)
#define NELEM ((size_t)Bn*Cn*Hn*Wn)

// --------------------------- CDNA5 async copy ------------------------------
// global_load_async_to_lds_b128: per-lane 16B global -> LDS, tracked by
// ASYNCcnt (GVS mode: 64-bit SGPR base + 32-bit VGPR byte offset).
__device__ __forceinline__ void async_cp_b128(unsigned lds_off,
                                              const h16* gbase,
                                              unsigned gbyteoff) {
  asm volatile("global_load_async_to_lds_b128 %0, %1, %2"
               :: "v"(lds_off), "v"(gbyteoff), "s"(gbase) : "memory");
}
__device__ __forceinline__ void wait_async0() {
  asm volatile("s_wait_asynccnt 0x0" ::: "memory");
}
// Low 32 bits of a generic pointer into LDS = wave-relative LDS byte offset.
__device__ __forceinline__ unsigned lds_addr(const void* p) {
  return (unsigned)(size_t)p;
}

// Load a 16x32 f16 WMMA A/B fragment row from LDS.
// Lane l: row = l&15, g = l>>4; halves [8g,8g+8) and [16+8g,16+8g+8).
__device__ __forceinline__ v16h frag_ld(const h16* __restrict__ row, int g) {
  v8h lo = *(const v8h*)(row + 8 * g);
  v8h hi = *(const v8h*)(row + 16 + 8 * g);
  v16h r;
#pragma unroll
  for (int i = 0; i < 8; ++i) { r[i] = lo[i]; r[8 + i] = hi[i]; }
  return r;
}

// ------------------------------ prep kernels -------------------------------

__global__ __launch_bounds__(256) void cvt_f32_f16(const float4* __restrict__ in,
                                                   v4h* __restrict__ out, int n4) {
  int i = blockIdx.x * 256 + threadIdx.x;
  if (i < n4) {
    float4 f = in[i];
    v4h v; v[0] = (h16)f.x; v[1] = (h16)f.y; v[2] = (h16)f.z; v[3] = (h16)f.w;
    out[i] = v;
  }
}

// src (Osrc, 96, 9) fp32 -> dst [Mpad][864] f16 with k = kk*96 + c (zero pad rows)
__global__ __launch_bounds__(256) void pack_w(const float* __restrict__ src,
                                              h16* __restrict__ dst,
                                              int Osrc, int Mpad) {
  int i = blockIdx.x * 256 + threadIdx.x;
  if (i >= Mpad * KDIM) return;
  int o = i / KDIM, k = i % KDIM;
  int kk = k / Cn, c = k % Cn;
  h16 v = (h16)0.f;
  if (o < Osrc) v = (h16)src[(o * Cn + c) * KKn + kk];
  dst[i] = v;
}

// ------------------------- conv1: dilated conv + fuse ----------------------
// grid = Bn * Hn * (Wn/64) = 1024, block = 256 (8 waves)

__global__ __launch_bounds__(256) void conv1_kernel(
    const float* __restrict__ x, const h16* __restrict__ xh,
    const h16* __restrict__ A1, const float* __restrict__ bias,
    float* __restrict__ x3f, h16* __restrict__ x3h) {
  __shared__ __align__(16) h16 As[2][96 * LDK];
  __shared__ __align__(16) h16 Bs[2][64 * LDK];

  int wg = blockIdx.x;
  int tx = (wg & 1) * 64;
  int y  = (wg >> 1) & (Hn - 1);
  int b  = wg >> 8;
  int tid = threadIdx.x;
  int lane = tid & 31, wave = tid >> 5;
  int g = lane >> 4, lm = lane & 15;

  const h16* xb = xh + (size_t)b * Cn * NPIX;

  auto stage = [&](int kc, int buf) {
    // A chunk: 96 rows x 32 halves = 384 x 16B async copies
    for (int e = tid; e < 384; e += 256) {
      int m = e >> 2, ko = (e & 3) << 3;
      async_cp_b128(lds_addr(&As[buf][m * LDK + ko]), A1,
                    (unsigned)((m * KDIM + kc * 32 + ko) * 2));
    }
    // B chunk: each thread owns 1 pixel x 8 contiguous k (channels)
    int kk = (kc * 32) / Cn, c0 = (kc * 32) % Cn;
    int dy = (kk / 3 - 1) * 2, dx = (kk % 3 - 1) * 2;   // dilation 2, pad 2
    int yy = y + dy;
    int p = tid & 63, kl0 = (tid >> 6) << 3;
    int xx = tx + p + dx;
    v8h v = {};
    if (yy >= 0 && yy < Hn && xx >= 0 && xx < Wn) {
      const h16* s = xb + ((size_t)(c0 + kl0) * Hn + yy) * Wn + xx;
#pragma unroll
      for (int i = 0; i < 8; ++i) v[i] = s[(size_t)i * NPIX];
    }
    *(v8h*)&Bs[buf][p * LDK + kl0] = v;
  };

  v8f acc0 = {}, acc1 = {}, acc2 = {};

  stage(0, 0);
  wait_async0();
  __syncthreads();
  for (int kc = 0; kc < KSTEPS; ++kc) {
    int cur = kc & 1;
    if (kc + 1 < KSTEPS) stage(kc + 1, cur ^ 1);
#pragma unroll
    for (int t = 0; t < 3; ++t) {
      int tile = wave + 8 * t;            // 0..23
      int mt = tile >> 2, nt = tile & 3;
      v16h af = frag_ld(&As[cur][(mt * 16 + lm) * LDK], g);
      v16h bf = frag_ld(&Bs[cur][(nt * 16 + lm) * LDK], g);
      v8f& acc = (t == 0) ? acc0 : ((t == 1) ? acc1 : acc2);
      acc = __builtin_amdgcn_wmma_f32_16x16x32_f16(false, af, false, bf,
                                                   (short)0, acc, false, false);
    }
    wait_async0();
    __syncthreads();
  }
  // epilogue: x3 = x + relu(conv + bias); emit fp32 and f16 copies
#pragma unroll
  for (int t = 0; t < 3; ++t) {
    int tile = wave + 8 * t;
    int mt = tile >> 2, nt = tile & 3;
    v8f acc = (t == 0) ? acc0 : ((t == 1) ? acc1 : acc2);
#pragma unroll
    for (int r = 0; r < 8; ++r) {
      int o = mt * 16 + r + 8 * g;
      int px = tx + nt * 16 + lm;
      size_t idx = (((size_t)b * Cn + o) * Hn + y) * Wn + px;
      float v = x[idx] + fmaxf(acc[r] + bias[o], 0.f);
      x3f[idx] = v;
      x3h[idx] = (h16)v;
    }
  }
}

// ----------------------- offset conv: 18 (pad 32) x 864 --------------------

__global__ __launch_bounds__(256) void offconv_kernel(
    const h16* __restrict__ x3h, const h16* __restrict__ Aw,
    const float* __restrict__ bias, float* __restrict__ off) {
  __shared__ __align__(16) h16 As[2][32 * LDK];
  __shared__ __align__(16) h16 Bs[2][64 * LDK];

  int wg = blockIdx.x;
  int tx = (wg & 1) * 64;
  int y  = (wg >> 1) & (Hn - 1);
  int b  = wg >> 8;
  int tid = threadIdx.x;
  int lane = tid & 31, wave = tid >> 5;
  int g = lane >> 4, lm = lane & 15;

  const h16* xb = x3h + (size_t)b * Cn * NPIX;

  auto stage = [&](int kc, int buf) {
    if (tid < 128) {                       // 32 rows x 4 chunks of 16B
      int m = tid >> 2, ko = (tid & 3) << 3;
      async_cp_b128(lds_addr(&As[buf][m * LDK + ko]), Aw,
                    (unsigned)((m * KDIM + kc * 32 + ko) * 2));
    }
    int kk = (kc * 32) / Cn, c0 = (kc * 32) % Cn;
    int dy = kk / 3 - 1, dx = kk % 3 - 1;               // pad 1
    int yy = y + dy;
    int p = tid & 63, kl0 = (tid >> 6) << 3;
    int xx = tx + p + dx;
    v8h v = {};
    if (yy >= 0 && yy < Hn && xx >= 0 && xx < Wn) {
      const h16* s = xb + ((size_t)(c0 + kl0) * Hn + yy) * Wn + xx;
#pragma unroll
      for (int i = 0; i < 8; ++i) v[i] = s[(size_t)i * NPIX];
    }
    *(v8h*)&Bs[buf][p * LDK + kl0] = v;
  };

  v8f acc = {};

  stage(0, 0);
  wait_async0();
  __syncthreads();
  for (int kc = 0; kc < KSTEPS; ++kc) {
    int cur = kc & 1;
    if (kc + 1 < KSTEPS) stage(kc + 1, cur ^ 1);
    int mt = wave >> 2, nt = wave & 3;                  // 8 tiles, 1 per wave
    v16h af = frag_ld(&As[cur][(mt * 16 + lm) * LDK], g);
    v16h bf = frag_ld(&Bs[cur][(nt * 16 + lm) * LDK], g);
    acc = __builtin_amdgcn_wmma_f32_16x16x32_f16(false, af, false, bf,
                                                 (short)0, acc, false, false);
    wait_async0();
    __syncthreads();
  }
  int mt = wave >> 2, nt = wave & 3;
#pragma unroll
  for (int r = 0; r < 8; ++r) {
    int o = mt * 16 + r + 8 * g;
    int px = tx + nt * 16 + lm;
    if (o < 18)
      off[(((size_t)b * 18 + o) * Hn + y) * Wn + px] = acc[r] + bias[o];
  }
}

// --------------- deformable sampling + 96x864 contraction ------------------
// Per (pixel,kk): 4 clamped tap offsets + 4 OOB-masked bilinear weights,
// so the B build is branch-free: 4 loads + 4 FMAs per element.

struct SampP { int o[4]; float w[4]; };

__global__ __launch_bounds__(256) void deform_kernel(
    const float* __restrict__ x3f, const float* __restrict__ off,
    const h16* __restrict__ A2, const float* __restrict__ bias,
    float* __restrict__ out) {
  __shared__ __align__(16) h16 As[2][96 * LDK];
  __shared__ __align__(16) h16 Bs[2][64 * LDK];
  __shared__ __align__(16) SampP P[64 * KKn];

  int wg = blockIdx.x;
  int tx = (wg & 1) * 64;
  int y  = (wg >> 1) & (Hn - 1);
  int b  = wg >> 8;
  int tid = threadIdx.x;
  int lane = tid & 31, wave = tid >> 5;
  int g = lane >> 4, lm = lane & 15;

  // per (pixel, kk) sampling parameters, computed once
  for (int e = tid; e < 64 * KKn; e += 256) {
    int p = e / KKn, kk = e % KKn;
    int px = tx + p;
    float dy = off[(((size_t)b * 18 + kk * 2 + 0) * Hn + y) * Wn + px];
    float dx = off[(((size_t)b * 18 + kk * 2 + 1) * Hn + y) * Wn + px];
    float py  = (float)y  + (float)(kk / 3 - 1) + dy;
    float pxx = (float)px + (float)(kk % 3 - 1) + dx;
    float fy = floorf(py), fx = floorf(pxx);
    int y0 = (int)fy, x0 = (int)fx;
    float wy = py - fy, wx = pxx - fx;
    SampP sp;
#pragma unroll
    for (int iy = 0; iy < 2; ++iy) {
#pragma unroll
      for (int ix = 0; ix < 2; ++ix) {
        int yy = y0 + iy, xx = x0 + ix;
        bool ok = (yy >= 0) & (yy < Hn) & (xx >= 0) & (xx < Wn);
        int yc = min(max(yy, 0), Hn - 1), xc = min(max(xx, 0), Wn - 1);
        float w = (iy ? wy : 1.f - wy) * (ix ? wx : 1.f - wx);
        sp.o[iy * 2 + ix] = yc * Wn + xc;
        sp.w[iy * 2 + ix] = ok ? w : 0.f;
      }
    }
    P[e] = sp;
  }
  __syncthreads();

  const float* xb = x3f + (size_t)b * Cn * NPIX;

  auto stage = [&](int kc, int buf) {
    for (int e = tid; e < 384; e += 256) {
      int m = e >> 2, ko = (e & 3) << 3;
      async_cp_b128(lds_addr(&As[buf][m * LDK + ko]), A2,
                    (unsigned)((m * KDIM + kc * 32 + ko) * 2));
    }
    int kk = (kc * 32) / Cn, c0 = (kc * 32) % Cn;
    int p = tid & 63, kl0 = (tid >> 6) << 3;
    SampP sp = P[p * KKn + kk];
    const float* base = xb + (size_t)(c0 + kl0) * NPIX;
    v8h v;
#pragma unroll
    for (int i = 0; i < 8; ++i) {
      const float* bc = base + (size_t)i * NPIX;
      float s = sp.w[0] * bc[sp.o[0]] + sp.w[1] * bc[sp.o[1]] +
                sp.w[2] * bc[sp.o[2]] + sp.w[3] * bc[sp.o[3]];
      v[i] = (h16)s;
    }
    *(v8h*)&Bs[buf][p * LDK + kl0] = v;
  };

  v8f acc0 = {}, acc1 = {}, acc2 = {};

  stage(0, 0);
  wait_async0();
  __syncthreads();
  for (int kc = 0; kc < KSTEPS; ++kc) {
    int cur = kc & 1;
    if (kc + 1 < KSTEPS) stage(kc + 1, cur ^ 1);
#pragma unroll
    for (int t = 0; t < 3; ++t) {
      int tile = wave + 8 * t;
      int mt = tile >> 2, nt = tile & 3;
      v16h af = frag_ld(&As[cur][(mt * 16 + lm) * LDK], g);
      v16h bf = frag_ld(&Bs[cur][(nt * 16 + lm) * LDK], g);
      v8f& acc = (t == 0) ? acc0 : ((t == 1) ? acc1 : acc2);
      acc = __builtin_amdgcn_wmma_f32_16x16x32_f16(false, af, false, bf,
                                                   (short)0, acc, false, false);
    }
    wait_async0();
    __syncthreads();
  }
#pragma unroll
  for (int t = 0; t < 3; ++t) {
    int tile = wave + 8 * t;
    int mt = tile >> 2, nt = tile & 3;
    v8f acc = (t == 0) ? acc0 : ((t == 1) ? acc1 : acc2);
#pragma unroll
    for (int r = 0; r < 8; ++r) {
      int o = mt * 16 + r + 8 * g;
      int px = tx + nt * 16 + lm;
      out[(((size_t)b * On + o) * Hn + y) * Wn + px] = acc[r] + bias[o];
    }
  }
}

// ------------------------------- launcher ----------------------------------

extern "C" void kernel_launch(void* const* d_in, const int* in_sizes, int n_in,
                              void* d_out, int out_size, void* d_ws, size_t ws_size,
                              hipStream_t stream) {
  (void)in_sizes; (void)n_in; (void)out_size; (void)ws_size;
  const float* x       = (const float*)d_in[0];
  const float* dconv_w = (const float*)d_in[1];
  const float* dconv_b = (const float*)d_in[2];
  const float* off_w   = (const float*)d_in[3];
  const float* off_b   = (const float*)d_in[4];
  const float* def_w   = (const float*)d_in[5];
  const float* def_b   = (const float*)d_in[6];
  float* out = (float*)d_out;

  char* ws = (char*)d_ws;
  size_t o = 0;
  auto alloc = [&](size_t bytes) -> void* {
    void* p = ws + o;
    o = (o + bytes + 255) & ~(size_t)255;
    return p;
  };
  h16*   xh   = (h16*)alloc(NELEM * 2);              // x in f16
  float* x3f  = (float*)alloc(NELEM * 4);            // x3 fp32
  h16*   x3h  = (h16*)alloc(NELEM * 2);              // x3 f16
  float* offb = (float*)alloc((size_t)Bn * 18 * NPIX * 4);
  h16*   A1   = (h16*)alloc((size_t)96 * KDIM * 2);
  h16*   Aoff = (h16*)alloc((size_t)32 * KDIM * 2);
  h16*   A2   = (h16*)alloc((size_t)96 * KDIM * 2);  // total ~55.5 MB

  int n4 = (int)(NELEM / 4);
  cvt_f32_f16<<<(n4 + 255) / 256, 256, 0, stream>>>((const float4*)x, (v4h*)xh, n4);
  pack_w<<<(96 * KDIM + 255) / 256, 256, 0, stream>>>(dconv_w, A1, 96, 96);
  pack_w<<<(32 * KDIM + 255) / 256, 256, 0, stream>>>(off_w, Aoff, 18, 32);
  pack_w<<<(96 * KDIM + 255) / 256, 256, 0, stream>>>(def_w, A2, 96, 96);

  dim3 grid(Bn * Hn * (Wn / 64));   // 1024
  conv1_kernel<<<grid, 256, 0, stream>>>(x, xh, A1, dconv_b, x3f, x3h);
  offconv_kernel<<<grid, 256, 0, stream>>>(x3h, Aoff, off_b, offb);
  deform_kernel<<<grid, 256, 0, stream>>>(x3f, offb, A2, def_b, out);
}
